// conv_layer_37598143709504
// MI455X (gfx1250) — compile-verified, matrix operand order checked by reading, not run on hardware
//
#include <hip/hip_runtime.h>
#include <hip/hip_bf16.h>
#include <stdint.h>

// CDNA5 WMMA vector types
typedef __attribute__((ext_vector_type(16))) __bf16 v16bf;
typedef __attribute__((ext_vector_type(8)))  float  v8f;

#define IN_F    32
#define POS     25
#define KTOT    800          // 25 * 32
#define OUT_F   64
#define MTILE   32           // nodes per block (2 x 16-row WMMA tiles)
#define ROW_PAD 808          // bf16 elems per LDS feat row (1616B: 16B aligned, conflict-free rows)

__device__ __forceinline__ unsigned short f32_bf16(float f) {
    union { float f; unsigned int u; } c; c.f = f;
    unsigned int r = c.u + 0x7FFFu + ((c.u >> 16) & 1u);   // round-to-nearest-even
    return (unsigned short)(r >> 16);
}

__global__ void w_to_bf16_kernel(const float* __restrict__ W,
                                 unsigned short* __restrict__ Wbf, int n) {
    int i = blockIdx.x * blockDim.x + threadIdx.x;
    if (i < n) Wbf[i] = f32_bf16(W[i]);
}

__global__ void __launch_bounds__(128)
spherical_conv_kernel(const float* __restrict__ x,
                      const int*   __restrict__ nidx,
                      const float* __restrict__ nw,
                      const unsigned short* __restrict__ Wbf,
                      const float* __restrict__ bias,
                      float* __restrict__ out,
                      int nodes)
{
    __shared__ unsigned short feat[MTILE][ROW_PAD];   // 32 nodes x 800 bf16 (padded)

    const int tid       = threadIdx.x;
    const int node_base = blockIdx.x * MTILE;

    // ---------------- Phase 1: gather + barycentric combine -> bf16 LDS tile ----
    // 32 nodes * 25 positions * 8 chunks(4 feats) = 6400 tasks; 128 threads * 50 iters
    for (int it = 0; it < 50; ++it) {
        int t    = it * 128 + tid;        // 0..6399
        int nloc = t / 200;               // 0..31
        int rem  = t - nloc * 200;
        int pos  = rem >> 3;              // 0..24
        int f0   = (rem & 7) * 4;         // 0,4,...,28

        int node  = node_base + nloc;
        int gnode = (node < nodes) ? node : (nodes - 1);   // clamp (tail block) - loads stay safe
        const int base = gnode * (POS * 3) + pos * 3;

        int   i0 = nidx[base + 0], i1 = nidx[base + 1], i2 = nidx[base + 2];
        float w0 = nw[base + 0],   w1 = nw[base + 1],   w2 = nw[base + 2];

        const float4 a0 = *(const float4*)(x + (long)i0 * IN_F + f0);
        const float4 a1 = *(const float4*)(x + (long)i1 * IN_F + f0);
        const float4 a2 = *(const float4*)(x + (long)i2 * IN_F + f0);

        float r0 = w0 * a0.x + w1 * a1.x + w2 * a2.x;
        float r1 = w0 * a0.y + w1 * a1.y + w2 * a2.y;
        float r2 = w0 * a0.z + w1 * a1.z + w2 * a2.z;
        float r3 = w0 * a0.w + w1 * a1.w + w2 * a2.w;

        uint2 packed;
        packed.x = (unsigned int)f32_bf16(r0) | ((unsigned int)f32_bf16(r1) << 16);
        packed.y = (unsigned int)f32_bf16(r2) | ((unsigned int)f32_bf16(r3) << 16);
        *(uint2*)&feat[nloc][pos * IN_F + f0] = packed;
    }
    __syncthreads();

    // ---------------- Phase 2: WMMA GEMM, M=32 (nodes) x N=64 x K=800 ------------
    // 4 waves; wave w owns output cols [w*16, w*16+16), both 16-row M tiles.
    // One B fragment per K-step feeds two WMMAs (halves W/L2 traffic per output).
    const int wave = tid >> 5;
    const int lane = tid & 31;
    const int half = lane >> 4;     // 0: K low 8 of each 16-group, 1: K high 8
    const int l15  = lane & 15;
    const int nb   = wave * 16 + l15;       // output feature (B column / C column)

    v8f acc0 = {};                           // rows node_base + 0..15
    v8f acc1 = {};                           // rows node_base + 16..31

    const unsigned short* wrow = Wbf + (long)nb * KTOT;   // W[nb][*], bf16, L2-resident

    #pragma unroll 5
    for (int kt = 0; kt < 25; ++kt) {
        const int koff = kt * 32 + half * 8;

        // B fragment: lane holds column n=nb; v16bf elems 0..7 = K koff..koff+7,
        // elems 8..15 = K koff+16..koff+23 (ISA 16-bit 16x32 packing).
        union { uint4 q[2]; v16bf v; } B;
        const unsigned short* bp = wrow + koff;
        B.q[0] = *(const uint4*)(bp);
        B.q[1] = *(const uint4*)(bp + 16);

        // A fragments from LDS: lane holds row m = l15 (tile0) / 16+l15 (tile1)
        union { uint4 q[2]; v16bf v; } A0, A1;
        const unsigned short* ap0 = &feat[l15][koff];
        A0.q[0] = *(const uint4*)(ap0);
        A0.q[1] = *(const uint4*)(ap0 + 16);
        const unsigned short* ap1 = &feat[16 + l15][koff];
        A1.q[0] = *(const uint4*)(ap1);
        A1.q[1] = *(const uint4*)(ap1 + 16);

        acc0 = __builtin_amdgcn_wmma_f32_16x16x32_bf16(
                   false, A0.v, false, B.v, (short)0, acc0, false, false);
        acc1 = __builtin_amdgcn_wmma_f32_16x16x32_bf16(
                   false, A1.v, false, B.v, (short)0, acc1, false, false);
    }

    // ---------------- Store: C layout VGPR r -> M = r + 8*half, N = l15 ----------
    const float bv = bias[nb];
    const int   m0 = half * 8;                       // first row of this lane's C slots
    float* outp = out + (long)(node_base + m0) * OUT_F + nb;

    if (node_base + MTILE <= nodes) {
        // fast path (all blocks except the last): uniform, unguarded stores
        #pragma unroll
        for (int r = 0; r < 8; ++r)
            outp[(long)r * OUT_F] = acc0[r] + bv;
        #pragma unroll
        for (int r = 0; r < 8; ++r)
            outp[(long)(16 + r) * OUT_F] = acc1[r] + bv;
    } else {
        #pragma unroll
        for (int r = 0; r < 8; ++r) {
            int node = node_base + m0 + r;
            if (node < nodes) out[(long)node * OUT_F + nb] = acc0[r] + bv;
        }
        #pragma unroll
        for (int r = 0; r < 8; ++r) {
            int node = node_base + 16 + m0 + r;
            if (node < nodes) out[(long)node * OUT_F + nb] = acc1[r] + bv;
        }
    }
}

extern "C" void kernel_launch(void* const* d_in, const int* in_sizes, int n_in,
                              void* d_out, int out_size, void* d_ws, size_t ws_size,
                              hipStream_t stream) {
    const float* x    = (const float*)d_in[0];
    const int*   nidx = (const int*)  d_in[1];
    const float* nw   = (const float*)d_in[2];
    const float* W    = (const float*)d_in[3];
    const float* b    = (const float*)d_in[4];
    float* out        = (float*)d_out;

    const int nodes = in_sizes[0] / IN_F;

    unsigned short* Wbf = (unsigned short*)d_ws;   // 64*800*2 = 102,400 bytes
    const int wn = OUT_F * KTOT;
    w_to_bf16_kernel<<<(wn + 255) / 256, 256, 0, stream>>>(W, Wbf, wn);

    const int blocks = (nodes + MTILE - 1) / MTILE;
    spherical_conv_kernel<<<blocks, 128, 0, stream>>>(x, nidx, nw, Wbf, b, out, nodes);
}
